// CrystalGraphConvNet_67293547594212
// MI455X (gfx1250) — matrix operand order repaired and checked by reading, not compiled, not run on hardware
//
#include <hip/hip_runtime.h>
#include <hip/hip_bf16.h>

#define NATOM 100000
#define MNBR  12
#define ORIGF 92
#define NBRF  41
#define AFEA  64
#define HFEA  128
#define NCONVL 3
#define NCRY  1000
#define KTOT  (2*AFEA + NBRF)   // 169
#define KPAD  192               // 6 k-steps of 32
#define KSTRIDE 200             // padded LDS row stride (bf16): 100 dwords, gcd-free
                                // vs 64 banks -> conflict-free B fragment loads
#define TWOA  128
#define EPSV  1e-5f

typedef __attribute__((ext_vector_type(16))) __bf16 v16bf;
typedef __attribute__((ext_vector_type(2)))  __bf16 v2bf;
typedef __attribute__((ext_vector_type(8)))  float  v8f;

static __device__ __forceinline__ unsigned short f2bf_sw(float f) {
  union { float f; unsigned u; } x; x.f = f;
  unsigned r = x.u + 0x7FFFu + ((x.u >> 16) & 1u);   // RNE
  return (unsigned short)(r >> 16);
}
static __device__ __forceinline__ v2bf cvt2bf(float lo, float hi) {
#if __has_builtin(__builtin_amdgcn_cvt_pk_bf16_f32)
  return __builtin_amdgcn_cvt_pk_bf16_f32(lo, hi);   // native v_cvt_pk_bf16_f32
#else
  union { v2bf v; unsigned short u[2]; } r;
  r.u[0] = f2bf_sw(lo); r.u[1] = f2bf_sw(hi);
  return r.v;
#endif
}
static __device__ __forceinline__ float softplusf(float x) {
  return fmaxf(x, 0.f) + log1pf(expf(-fabsf(x)));
}
static __device__ __forceinline__ float sigmoidf(float x) {
  return 1.f / (1.f + expf(-x));
}

// ---------------- small helper kernels ----------------
__global__ void k_zero(float* p, int n) {
  int i = blockIdx.x * blockDim.x + threadIdx.x;
  if (i < n) p[i] = 0.f;
}

__global__ void k_embed(const float* __restrict__ atom_fea,
                        const float* __restrict__ embW,
                        const float* __restrict__ embB,
                        float* __restrict__ x) {
  int idx = blockIdx.x * blockDim.x + threadIdx.x;
  if (idx >= NATOM * AFEA) return;
  int n = idx >> 6, c = idx & 63;
  float acc = embB[c];
  #pragma unroll 4
  for (int k = 0; k < ORIGF; ++k)
    acc += atom_fea[n * ORIGF + k] * embW[k * AFEA + c];
  x[idx] = acc;
}

// ---------------- fused conv GEMM (WMMA bf16) ----------------
// One atom per wave-tile: A = 16 rows (12 nbrs + 4 zero pad) x 192 K (bf16).
// PASS 1: raw sum / sumsq of C over the N*M valid rows (bias & padding cancel
//         analytically in the finalize step).
// PASS 2: BN1 affine, sigmoid(filter)*softplus(core), reduce over M,
//         write nbr_sumed and accumulate BN2 stats.
template <int PASS>
__global__ __launch_bounds__(256)
__attribute__((amdgpu_waves_per_eu(4)))
void k_conv(
    const float* __restrict__ x,
    const float* __restrict__ nbr_fea,
    const int*   __restrict__ nbr_idx,
    const float* __restrict__ W,        // (169,128) row-major for this layer
    const float* __restrict__ scale1,   // 128   (pass 2)
    const float* __restrict__ shift1,   // 128   (pass 2)
    float* __restrict__ stats1,         // 256   (pass 1 out: sumC, sumC^2)
    float* __restrict__ nbr_sumed,      // N*64  (pass 2 out)
    float* __restrict__ stats2)         // 128   (pass 2 out: sum, sumsq)
{
  __shared__ union { unsigned short u[TWOA * KSTRIDE]; uint4 q[TWOA * KSTRIDE / 8]; } sW;
  __shared__ float redA[TWOA];
  __shared__ float redB[TWOA];

  // Stage W transposed + zero-padded to K=192 as bf16 in LDS: sW[ch][k],
  // row stride padded to 200 (tail 192..199 never read).
  for (int i = threadIdx.x; i < TWOA * KPAD; i += blockDim.x) {
    int ch = i / KPAD, k = i - ch * KPAD;
    float w = (k < KTOT) ? W[k * TWOA + ch] : 0.f;
    union { v2bf v; unsigned short u[2]; } cv;
    cv.v = cvt2bf(w, 0.f);
    sW.u[ch * KSTRIDE + k] = cv.u[0];
  }
  if (threadIdx.x < TWOA) { redA[threadIdx.x] = 0.f; redB[threadIdx.x] = 0.f; }
  __syncthreads();

  const int lane   = threadIdx.x & 31;
  const int l15    = lane & 15;          // A row (=m) during build; B/C column in tile
  const int khalf  = lane >> 4;          // 0 or 1
  const int wid    = blockIdx.x * (blockDim.x >> 5) + (threadIdx.x >> 5);
  const int nwaves = gridDim.x * (blockDim.x >> 5);

  // hoisted per-lane channel constants (pass 2)
  float sc[8], sh[8];
  #pragma unroll
  for (int t = 0; t < 8; ++t) {
    int c = t * 16 + l15;
    sc[t] = (PASS == 2) ? scale1[c] : 0.f;
    sh[t] = (PASS == 2) ? shift1[c] : 0.f;
  }

  float accS[8], accQ[8];
  float a2S[4] = {0.f, 0.f, 0.f, 0.f}, a2Q[4] = {0.f, 0.f, 0.f, 0.f};
  #pragma unroll
  for (int t = 0; t < 8; ++t) { accS[t] = 0.f; accQ[t] = 0.f; }

  union AF { v16bf v; v2bf p[8]; unsigned d[8]; };
  union BF { v16bf v; uint4 q[2]; };

  for (int n = wid; n < NATOM; n += nwaves) {
    const bool rowv = (l15 < MNBR);
    int nbr = 0;
    if (rowv) nbr = nbr_idx[n * MNBR + l15];
    const float* xn = x + (size_t)n  * AFEA;
    const float* xb = x + (size_t)nbr * AFEA;
    const float* nf = nbr_fea + ((size_t)n * MNBR + l15) * NBRF;

    // ---- build all 6 A fragments for this atom (contiguous k order, same as loadB:
    // lane row = l15, dwords 0-7 hold K kb..kb+15, kb = ks*32 + khalf*16)
    AF A[6];
    #pragma unroll
    for (int ks = 0; ks < 6; ++ks) {
      float e[16];
      #pragma unroll
      for (int c = 0; c < 2; ++c) {
        const int kc = ks * 32 + khalf * 16 + c * 8;   // chunk start (mult of 8)
        float* d = e + c * 8;
        if (kc < 2 * AFEA) {
          // x rows are 64 floats -> 32B-aligned chunks: two B128 loads
          const float* src = (kc < AFEA) ? (xn + kc) : (xb + (kc - AFEA));
          float4 q0 = ((const float4*)src)[0];
          float4 q1 = ((const float4*)src)[1];
          d[0] = q0.x; d[1] = q0.y; d[2] = q0.z; d[3] = q0.w;
          d[4] = q1.x; d[5] = q1.y; d[6] = q1.z; d[7] = q1.w;
        } else {
          // nbr_fea rows are 41 floats (4B aligned): scalar, bounds-checked
          #pragma unroll
          for (int i = 0; i < 8; ++i) {
            int k = kc + i;
            d[i] = (k < KTOT) ? nf[k - 2 * AFEA] : 0.f;
          }
        }
      }
      #pragma unroll
      for (int i = 0; i < 8; ++i)
        A[ks].p[i] = cvt2bf(e[2 * i], e[2 * i + 1]);
      #pragma unroll
      for (int i = 0; i < 8; ++i)
        A[ks].d[i] = rowv ? A[ks].d[i] : 0u;           // zero the 4 pad rows
    }

    // Launder the LDS base so the per-tile B loads cannot be LICM-hoisted out
    // of the atom loop (hoisting 48 fragments = 384 VGPRs -> scratch spills).
    // The ds_load stays adjacent to its WMMA; offsets still fold to immediates.
    unsigned wbase = 0;
    asm volatile("" : "+v"(wbase));
    const unsigned short* wu = sW.u + wbase;

    // B fragment loader: column = t*16+l15, K = ks*32 + khalf*16 .. +15
    // (two 16B-aligned dwordx4 LDS loads; conflict-free thanks to KSTRIDE)
    auto loadB = [&](int t, int ks) -> v16bf {
      const uint4* bp = (const uint4*)(wu + (t * 16 + l15) * KSTRIDE
                                          + ks * 32 + khalf * 16);
      BF b; b.q[0] = bp[0]; b.q[1] = bp[1];
      return b.v;
    };

    if (PASS == 1) {
      // column tiles one at a time: only one 8-VGPR accumulator live
      #pragma unroll
      for (int t = 0; t < 8; ++t) {
        v8f C;
        #pragma unroll
        for (int e = 0; e < 8; ++e) C[e] = 0.f;
        #pragma unroll
        for (int ks = 0; ks < 6; ++ks) {
          v16bf bfr = loadB(t, ks);
          C = __builtin_amdgcn_wmma_f32_16x16x32_bf16(
                  false, A[ks].v, false, bfr, (short)0, C, false, false);
        }
        // pad rows give C==0 exactly -> accumulate unconditionally
        float s = 0.f, q = 0.f;
        #pragma unroll
        for (int v = 0; v < 8; ++v) { s += C[v]; q = fmaf(C[v], C[v], q); }
        accS[t] += s; accQ[t] += q;
      }
    } else {
      #pragma unroll
      for (int t = 0; t < 4; ++t) {
        v8f Cf, Cc;
        #pragma unroll
        for (int e = 0; e < 8; ++e) { Cf[e] = 0.f; Cc[e] = 0.f; }
        #pragma unroll
        for (int ks = 0; ks < 6; ++ks) {
          v16bf bf0 = loadB(t, ks);
          Cf = __builtin_amdgcn_wmma_f32_16x16x32_bf16(
                   false, A[ks].v, false, bf0, (short)0, Cf, false, false);
        }
        #pragma unroll
        for (int ks = 0; ks < 6; ++ks) {
          v16bf bf1 = loadB(t + 4, ks);
          Cc = __builtin_amdgcn_wmma_f32_16x16x32_bf16(
                   false, A[ks].v, false, bf1, (short)0, Cc, false, false);
        }
        // rows m = v + khalf*8; v<4 always valid, v>=4 valid iff khalf==0
        const float msk = khalf ? 0.f : 1.f;
        float part = 0.f;
        #pragma unroll
        for (int v = 0; v < 8; ++v) {
          float gf = Cf[v] * sc[t]     + sh[t];
          float gc = Cc[v] * sc[t + 4] + sh[t + 4];
          float p  = sigmoidf(gf) * softplusf(gc);
          part += (v < 4) ? p : p * msk;
        }
        float tot = part + __shfl_xor(part, 16, 32);   // m 0-7 + m 8-15
        if (lane < 16) {
          nbr_sumed[(size_t)n * AFEA + t * 16 + l15] = tot;
          a2S[t] += tot; a2Q[t] = fmaf(tot, tot, a2Q[t]);
        }
      }
    }
  }

  // block-level LDS reduction, then one global atomic per channel per block
  if (PASS == 1) {
    #pragma unroll
    for (int t = 0; t < 8; ++t) {
      atomicAdd(&redA[t * 16 + l15], accS[t]);
      atomicAdd(&redB[t * 16 + l15], accQ[t]);
    }
    __syncthreads();
    if (threadIdx.x < TWOA) {
      atomicAdd(&stats1[threadIdx.x],        redA[threadIdx.x]);
      atomicAdd(&stats1[TWOA + threadIdx.x], redB[threadIdx.x]);
    }
  } else {
    if (lane < 16)
      #pragma unroll
      for (int t = 0; t < 4; ++t) {
        atomicAdd(&redA[t * 16 + l15], a2S[t]);
        atomicAdd(&redB[t * 16 + l15], a2Q[t]);
      }
    __syncthreads();
    if (threadIdx.x < AFEA) {
      atomicAdd(&stats2[threadIdx.x],        redA[threadIdx.x]);
      atomicAdd(&stats2[AFEA + threadIdx.x], redB[threadIdx.x]);
    }
  }
}

// ---------------- BN finalize kernels ----------------
// BN1 on (z + bias): mean = meanC + bias, var = var(C); bias cancels:
// out = z*s + (b1 - meanC*s)
__global__ void k_fin1(const float* __restrict__ stats1,
                       const float* __restrict__ g1, const float* __restrict__ b1,
                       float* __restrict__ scale1, float* __restrict__ shift1) {
  int c = threadIdx.x;                       // 128 threads
  const float CNT = (float)NATOM * (float)MNBR;
  float meanC = stats1[c] / CNT;
  float var   = stats1[TWOA + c] / CNT - meanC * meanC;
  float s     = rsqrtf(var + EPSV) * g1[c];
  scale1[c] = s;
  shift1[c] = b1[c] - meanC * s;
}

__global__ void k_fin2(const float* __restrict__ stats2,
                       const float* __restrict__ g2, const float* __restrict__ b2,
                       float* __restrict__ s2, float* __restrict__ t2) {
  int c = threadIdx.x;                       // 64 threads
  const float CNT = (float)NATOM;
  float mean = stats2[c] / CNT;
  float var  = stats2[AFEA + c] / CNT - mean * mean;
  float s    = rsqrtf(var + EPSV) * g2[c];
  s2[c] = s;
  t2[c] = b2[c] - mean * s;
}

__global__ void k_update(const float* __restrict__ xc,
                         const float* __restrict__ ns,
                         const float* __restrict__ s2, const float* __restrict__ t2,
                         float* __restrict__ xn) {
  int idx = blockIdx.x * blockDim.x + threadIdx.x;
  if (idx >= NATOM * AFEA) return;
  int c = idx & 63;
  xn[idx] = softplusf(xc[idx] + ns[idx] * s2[c] + t2[c]);
}

// ---------------- pooling + head ----------------
__global__ void k_pool(const float* __restrict__ x, const int* __restrict__ cry,
                       float* __restrict__ sums, float* __restrict__ cnts) {
  int idx = blockIdx.x * blockDim.x + threadIdx.x;
  if (idx >= NATOM * AFEA) return;
  int n = idx >> 6, c = idx & 63;
  int g = cry[n];
  atomicAdd(&sums[g * AFEA + c], x[idx]);
  if (c == 0) atomicAdd(&cnts[g], 1.f);
}

__global__ __launch_bounds__(128) void k_head(
    const float* __restrict__ sums, const float* __restrict__ cnts,
    const float* __restrict__ fcW, const float* __restrict__ fcb,
    const float* __restrict__ outW, const float* __restrict__ outb,
    float* __restrict__ out) {
  __shared__ float sp[AFEA];
  __shared__ float red[HFEA];
  int g = blockIdx.x, j = threadIdx.x;
  if (j < AFEA) sp[j] = softplusf(sums[g * AFEA + j] / cnts[g]);
  __syncthreads();
  float h = fcb[j];
  #pragma unroll 4
  for (int c = 0; c < AFEA; ++c) h += sp[c] * fcW[c * HFEA + j];
  h = softplusf(h);
  red[j] = h * outW[j];
  __syncthreads();
  for (int s = 64; s > 0; s >>= 1) {
    if (j < s) red[j] += red[j + s];
    __syncthreads();
  }
  if (j == 0) out[g] = red[0] + outb[0];
}

// ---------------- host orchestration ----------------
extern "C" void kernel_launch(void* const* d_in, const int* in_sizes, int n_in,
                              void* d_out, int out_size, void* d_ws, size_t ws_size,
                              hipStream_t stream) {
  const float* atom_fea = (const float*)d_in[0];
  const float* nbr_fea  = (const float*)d_in[1];
  const int*   nbr_idx  = (const int*)d_in[2];
  const int*   cry_idx  = (const int*)d_in[3];
  const float* emb_W    = (const float*)d_in[4];
  const float* emb_b    = (const float*)d_in[5];
  const float* conv_W   = (const float*)d_in[6];
  const float* conv_b   = (const float*)d_in[7];   // cancels inside BN1
  const float* bn1_g    = (const float*)d_in[8];
  const float* bn1_b    = (const float*)d_in[9];
  const float* bn2_g    = (const float*)d_in[10];
  const float* bn2_b    = (const float*)d_in[11];
  const float* fc_W     = (const float*)d_in[12];
  const float* fc_b     = (const float*)d_in[13];
  const float* out_W    = (const float*)d_in[14];
  const float* out_b    = (const float*)d_in[15];
  (void)conv_b;
  float* out = (float*)d_out;

  // workspace carve-up (~77 MB)
  char*  w   = (char*)d_ws;
  size_t off = 0;
  auto carve = [&](size_t bytes) {
    void* p = w + off;
    off += (bytes + 255) & ~(size_t)255;
    return p;
  };
  float* xA        = (float*)carve((size_t)NATOM * AFEA * 4);
  float* xB        = (float*)carve((size_t)NATOM * AFEA * 4);
  float* nbr_sumed = (float*)carve((size_t)NATOM * AFEA * 4);
  float* stats1    = (float*)carve(2 * TWOA * 4);
  float* stats2    = (float*)carve(2 * AFEA * 4);
  float* scale1    = (float*)carve(TWOA * 4);
  float* shift1    = (float*)carve(TWOA * 4);
  float* s2        = (float*)carve(AFEA * 4);
  float* t2        = (float*)carve(AFEA * 4);
  float* psum      = (float*)carve((size_t)NCRY * AFEA * 4);
  float* pcnt      = (float*)carve((size_t)NCRY * 4);

  const int NTHREADS  = 256;
  const int NC_GRID   = (NATOM * AFEA + NTHREADS - 1) / NTHREADS;
  const int CONV_GRID = 640;   // 5120 waves, grid-stride over 100k atom-tiles

  k_embed<<<NC_GRID, NTHREADS, 0, stream>>>(atom_fea, emb_W, emb_b, xA);

  float* xcur = xA;
  float* xnxt = xB;
  for (int l = 0; l < NCONVL; ++l) {
    const float* Wl = conv_W + (size_t)l * KTOT * TWOA;
    k_zero<<<1, 256, 0, stream>>>(stats1, 2 * TWOA);
    k_zero<<<1, 128, 0, stream>>>(stats2, 2 * AFEA);
    k_conv<1><<<CONV_GRID, NTHREADS, 0, stream>>>(
        xcur, nbr_fea, nbr_idx, Wl, nullptr, nullptr,
        stats1, nullptr, nullptr);
    k_fin1<<<1, TWOA, 0, stream>>>(stats1,
        bn1_g + (size_t)l * TWOA, bn1_b + (size_t)l * TWOA, scale1, shift1);
    k_conv<2><<<CONV_GRID, NTHREADS, 0, stream>>>(
        xcur, nbr_fea, nbr_idx, Wl, scale1, shift1,
        nullptr, nbr_sumed, stats2);
    k_fin2<<<1, AFEA, 0, stream>>>(stats2,
        bn2_g + (size_t)l * AFEA, bn2_b + (size_t)l * AFEA, s2, t2);
    k_update<<<NC_GRID, NTHREADS, 0, stream>>>(xcur, nbr_sumed, s2, t2, xnxt);
    float* tmp = xcur; xcur = xnxt; xnxt = tmp;
  }

  k_zero<<<(NCRY * AFEA + 255) / 256, 256, 0, stream>>>(psum, NCRY * AFEA);
  k_zero<<<(NCRY + 255) / 256, 256, 0, stream>>>(pcnt, NCRY);
  k_pool<<<NC_GRID, NTHREADS, 0, stream>>>(xcur, cry_idx, psum, pcnt);
  k_head<<<NCRY, HFEA, 0, stream>>>(psum, pcnt, fc_W, fc_b, out_W, out_b, out);
}
